// SparseMoEBlock_9328668967122
// MI455X (gfx1250) — matrix operand order repaired and checked by reading, not compiled, not run on hardware
//
#include <hip/hip_runtime.h>
#include <hip/hip_bf16.h>

typedef __attribute__((ext_vector_type(16))) __bf16 v16bf;
typedef __attribute__((ext_vector_type(8)))  __bf16 v8bf;
typedef __attribute__((ext_vector_type(8)))  float  v8f;
typedef __attribute__((ext_vector_type(4)))  float  v4f;
typedef __attribute__((ext_vector_type(2)))  float  v2f;
typedef unsigned int u32x4 __attribute__((ext_vector_type(4)));
typedef int          i32x4 __attribute__((ext_vector_type(4)));
typedef int          i32x8 __attribute__((ext_vector_type(8)));

#define NTOK 4096
#define DDIM 1024
#define HDIM 4096
#define EEXP 8
#define KSEL 8192   /* B*L*capacity = 4096*2 */

#if defined(__has_builtin)
# if __has_builtin(__builtin_amdgcn_tensor_load_to_lds)
#  define HAVE_TDM 1
# endif
#endif
#ifndef HAVE_TDM
# define HAVE_TDM 0
#endif

__device__ __forceinline__ float gelu_exact(float x) {
    return 0.5f * x * (1.0f + erff(x * 0.70710678118654752f));
}
__device__ __forceinline__ float gelu_tanh(float x) {
    float x3 = x * x * x;
    return 0.5f * x * (1.0f + tanhf(0.7978845608028654f * (x + 0.044715f * x3)));
}
__device__ __forceinline__ float siluf(float x) { return x / (1.0f + expf(-x)); }

#if HAVE_TDM
// TDM 2D tile load: tile_d1 rows x tile_d0 elems (bf16) from row-major tensor
// with row stride stride_elems, into LDS at lds_addr. D# per CDNA5 ISA §8.3/8.4.
__device__ __forceinline__ void tdm_load_2d_bf16(unsigned lds_addr,
                                                 unsigned long long gaddr,
                                                 unsigned tensor_d0, unsigned tensor_d1,
                                                 unsigned tile_d0, unsigned tile_d1,
                                                 unsigned stride_elems) {
    u32x4 g0;
    g0[0] = 1u;                                           // count=1 (valid), user mode
    g0[1] = lds_addr;                                     // lds_addr [63:32]
    g0[2] = (unsigned)(gaddr & 0xFFFFFFFFu);              // global_addr [95:64]
    g0[3] = (unsigned)((gaddr >> 32) & 0x1FFFFFFu) | (2u << 30);  // addr hi + type=2
    i32x8 g1;
    g1[0] = (int)(1u << 16);                              // wg_mask=0, data_size=1 (2B)
    g1[1] = (int)((tensor_d0 & 0xFFFFu) << 16);           // tensor_dim0[15:0] at [63:48]
    g1[2] = (int)((tensor_d0 >> 16) | ((tensor_d1 & 0xFFFFu) << 16));
    g1[3] = (int)((tensor_d1 >> 16) | (tile_d0 << 16));   // tile_dim0 at [127:112]
    g1[4] = (int)(tile_d1 & 0xFFFFu);                     // tile_dim1; tile_dim2=0
    g1[5] = (int)stride_elems;                            // tensor_dim0_stride[31:0]
    g1[6] = 0;
    g1[7] = 0;
    i32x4 z4 = {0, 0, 0, 0};
#if defined(__clang_major__) && (__clang_major__ >= 23)
    i32x8 z8 = {0, 0, 0, 0, 0, 0, 0, 0};
    __builtin_amdgcn_tensor_load_to_lds(g0, g1, z4, z4, z8, 0);
#else
    __builtin_amdgcn_tensor_load_to_lds(g0, g1, z4, z4, 0);
#endif
}
#endif

// ---------------------------------------------------------------- converts
__global__ void k_cvt_bf16(const float* __restrict__ in, __bf16* __restrict__ out, int n) {
    int i = blockIdx.x * blockDim.x + threadIdx.x;
    if (i < n) out[i] = (__bf16)in[i];
}

// w[R][C] fp32 -> wt[C][R] bf16 (K-contiguous per output row for WMMA B operand)
__global__ void k_cvtT_bf16(const float* __restrict__ in, __bf16* __restrict__ out,
                            int R, int C) {
    __shared__ float tile[32][33];
    int cb = blockIdx.x * 32, rb = blockIdx.y * 32;
    int tx = threadIdx.x, ty = threadIdx.y;   // blockDim (32,8)
#pragma unroll
    for (int j = 0; j < 32; j += 8)
        tile[ty + j][tx] = in[(size_t)(rb + ty + j) * C + cb + tx];
    __syncthreads();
#pragma unroll
    for (int j = 0; j < 32; j += 8)
        out[(size_t)(cb + ty + j) * R + rb + tx] = (__bf16)tile[tx][ty + j];
}

__global__ void k_zero(float* __restrict__ p, int n) {
    int i = blockIdx.x * blockDim.x + threadIdx.x;
    if (i < n) p[i] = 0.0f;
}

// ------------------------------------------ fp32 tensor-core GEMM + activation
// C = act(A[M,K] @ B[K,N] + bias), exact fp32 via V_WMMA_F32_16X16X4_F32.
template <int ACT>
__global__ void __launch_bounds__(256)
k_gemm_f32_act(const float* __restrict__ A, const float* __restrict__ Bw,
               const float* __restrict__ bias, float* __restrict__ Cout,
               int M, int N, int K) {
    __shared__ float As[64 * 16];
    __shared__ float Bs[16 * 64];
    int tid = threadIdx.x;
    int wave = tid >> 5, lane = tid & 31;
    int wm = wave & 3, wn = wave >> 2;          // 4 x 2 wave grid
    int bm = blockIdx.y * 64, bn = blockIdx.x * 64;
    int l15 = lane & 15, hi = lane >> 4;
    v8f acc[2] = {};
    for (int k0 = 0; k0 < K; k0 += 16) {
        int idx = tid * 4;
        int ra = idx >> 4, ca = idx & 15;
        *(v4f*)&As[idx] = *(const v4f*)&A[(size_t)(bm + ra) * K + k0 + ca];
        int rb = idx >> 6, cbv = idx & 63;
        *(v4f*)&Bs[idx] = *(const v4f*)&Bw[(size_t)(k0 + rb) * N + bn + cbv];
        __syncthreads();
        int kb = hi * 2;  // lanes 16-31 hold K=2,3 of each 4-step
#pragma unroll
        for (int kk = 0; kk < 16; kk += 4) {
            v2f a = *(const v2f*)&As[(wm * 16 + l15) * 16 + kk + kb];
#pragma unroll
            for (int ni = 0; ni < 2; ni++) {
                int n = wn * 32 + ni * 16 + l15;
                v2f b;
                b.x = Bs[(kk + kb) * 64 + n];
                b.y = Bs[(kk + kb + 1) * 64 + n];
                acc[ni] = __builtin_amdgcn_wmma_f32_16x16x4_f32(
                    false, a, false, b, (short)0, acc[ni], false, false);
            }
        }
        __syncthreads();
    }
#pragma unroll
    for (int ni = 0; ni < 2; ni++) {
        int col = bn + wn * 32 + ni * 16 + l15;
        float bv = bias[col];
#pragma unroll
        for (int r = 0; r < 8; r++) {
            int row = bm + wm * 16 + r + hi * 8;
            float v = acc[ni][r] + bv;
            Cout[(size_t)row * N + col] = (ACT == 0) ? gelu_exact(v) : siluf(v);
        }
    }
}

// --------------------------------------------------- bf16 WMMA GEMM core
// BM=BN=128, BK=32, 256 threads = 8 waves, wave tile 32x64 (2x4 16x16 frags).
// Tiles staged into double-buffered LDS by the Tensor Data Mover; wave 0 issues
// tile i+1 while all waves run WMMA on tile i (TENSORcnt<=2 => tile i landed).
#define FFN_TILE_ELEMS (128 * 32)
#define FFN_TILE_BYTES (FFN_TILE_ELEMS * 2)

#define FFN_ACCUM_LOOP(AS_PTR, BS_PTR)                                          \
    {                                                                           \
        int koA = hi * 16, koB = hi * 16;                                       \
        v16bf afr[2], bfr[4];                                                   \
        _Pragma("unroll")                                                       \
        for (int mi = 0; mi < 2; mi++) {                                        \
            int m = wm * 32 + mi * 16 + l15;                                    \
            union { v16bf v; v8bf h[2]; } u;                                    \
            u.h[0] = *(const v8bf*)&(AS_PTR)[m * 32 + koA];                     \
            u.h[1] = *(const v8bf*)&(AS_PTR)[m * 32 + koA + 8];                 \
            afr[mi] = u.v;                                                      \
        }                                                                       \
        _Pragma("unroll")                                                       \
        for (int ni = 0; ni < 4; ni++) {                                        \
            int n = wn * 64 + ni * 16 + l15;                                    \
            union { v16bf v; v8bf h[2]; } u;                                    \
            u.h[0] = *(const v8bf*)&(BS_PTR)[n * 32 + koB];                     \
            u.h[1] = *(const v8bf*)&(BS_PTR)[n * 32 + koB + 8];                 \
            bfr[ni] = u.v;                                                      \
        }                                                                       \
        _Pragma("unroll")                                                       \
        for (int mi = 0; mi < 2; mi++)                                          \
            _Pragma("unroll")                                                   \
            for (int ni = 0; ni < 4; ni++)                                      \
                acc[mi][ni] = __builtin_amdgcn_wmma_f32_16x16x32_bf16(          \
                    false, afr[mi], false, bfr[ni], (short)0, acc[mi][ni],      \
                    false, false);                                              \
    }

__global__ void __launch_bounds__(256)
k_gemm_bf16_ffn1(const __bf16* __restrict__ A, const __bf16* __restrict__ Bt,
                 const float* __restrict__ bias, __bf16* __restrict__ Hout,
                 int M, int N, int K) {
    __shared__ __bf16 As[2][FFN_TILE_ELEMS];
    __shared__ __bf16 Bs[2][FFN_TILE_ELEMS];
    int tid = threadIdx.x, wave = tid >> 5, lane = tid & 31;
    int wm = wave >> 1, wn = wave & 1;
    int bm = blockIdx.y * 128, bn = blockIdx.x * 128;
    int l15 = lane & 15, hi = lane >> 4;
    v8f acc[2][4] = {};
#if HAVE_TDM
    unsigned ldsA0 = (unsigned)(size_t)&As[0][0];
    unsigned ldsB0 = (unsigned)(size_t)&Bs[0][0];
    int niter = K >> 5;
    if (wave == 0) {
        tdm_load_2d_bf16(ldsA0, (unsigned long long)(size_t)(A + (size_t)bm * K),
                         (unsigned)K, (unsigned)M, 32u, 128u, (unsigned)K);
        tdm_load_2d_bf16(ldsB0, (unsigned long long)(size_t)(Bt + (size_t)bn * K),
                         (unsigned)K, (unsigned)N, 32u, 128u, (unsigned)K);
    }
    for (int i = 0; i < niter; i++) {
        int cur = i & 1, nxt = cur ^ 1;
        if (wave == 0) {
            if (i + 1 < niter) {
                int k1 = (i + 1) << 5;
                tdm_load_2d_bf16(ldsA0 + nxt * FFN_TILE_BYTES,
                                 (unsigned long long)(size_t)(A + (size_t)bm * K + k1),
                                 (unsigned)K, (unsigned)M, 32u, 128u, (unsigned)K);
                tdm_load_2d_bf16(ldsB0 + nxt * FFN_TILE_BYTES,
                                 (unsigned long long)(size_t)(Bt + (size_t)bn * K + k1),
                                 (unsigned)K, (unsigned)N, 32u, 128u, (unsigned)K);
                __builtin_amdgcn_s_wait_tensorcnt(2);  // tile i landed; i+1 in flight
            } else {
                __builtin_amdgcn_s_wait_tensorcnt(0);
            }
        }
        __syncthreads();
        const __bf16* asp = &As[cur][0];
        const __bf16* bsp = &Bs[cur][0];
        FFN_ACCUM_LOOP(asp, bsp)
        __syncthreads();   // done reading buf[cur] before it is overwritten
    }
#else
    for (int k0 = 0; k0 < K; k0 += 32) {
#pragma unroll
        for (int i = 0; i < 2; i++) {
            int c = tid + 256 * i;
            int r = c >> 2, cc = (c & 3) * 8;
            *(v8bf*)&As[0][r * 32 + cc] = *(const v8bf*)&A[(size_t)(bm + r) * K + k0 + cc];
            *(v8bf*)&Bs[0][r * 32 + cc] = *(const v8bf*)&Bt[(size_t)(bn + r) * K + k0 + cc];
        }
        __syncthreads();
        const __bf16* asp = &As[0][0];
        const __bf16* bsp = &Bs[0][0];
        FFN_ACCUM_LOOP(asp, bsp)
        __syncthreads();
    }
#endif
#pragma unroll
    for (int ni = 0; ni < 4; ni++) {
        int col = bn + wn * 64 + ni * 16 + l15;
        float bv = bias[col];
#pragma unroll
        for (int mi = 0; mi < 2; mi++)
#pragma unroll
            for (int r = 0; r < 8; r++) {
                int row = bm + wm * 32 + mi * 16 + r + hi * 8;
                Hout[(size_t)row * N + col] = (__bf16)gelu_tanh(acc[mi][ni][r] + bv);
            }
    }
}

__global__ void __launch_bounds__(256)
k_gemm_bf16_ffn2(const __bf16* __restrict__ A, const __bf16* __restrict__ Bt,
                 const float* __restrict__ bias, const float* __restrict__ wgt,
                 float* __restrict__ Out, int M, int N, int K) {
    __shared__ __bf16 As[2][FFN_TILE_ELEMS];
    __shared__ __bf16 Bs[2][FFN_TILE_ELEMS];
    int tid = threadIdx.x, wave = tid >> 5, lane = tid & 31;
    int wm = wave >> 1, wn = wave & 1;
    int bm = blockIdx.y * 128, bn = blockIdx.x * 128;
    int l15 = lane & 15, hi = lane >> 4;
    v8f acc[2][4] = {};
#if HAVE_TDM
    unsigned ldsA0 = (unsigned)(size_t)&As[0][0];
    unsigned ldsB0 = (unsigned)(size_t)&Bs[0][0];
    int niter = K >> 5;
    if (wave == 0) {
        tdm_load_2d_bf16(ldsA0, (unsigned long long)(size_t)(A + (size_t)bm * K),
                         (unsigned)K, (unsigned)M, 32u, 128u, (unsigned)K);
        tdm_load_2d_bf16(ldsB0, (unsigned long long)(size_t)(Bt + (size_t)bn * K),
                         (unsigned)K, (unsigned)N, 32u, 128u, (unsigned)K);
    }
    for (int i = 0; i < niter; i++) {
        int cur = i & 1, nxt = cur ^ 1;
        if (wave == 0) {
            if (i + 1 < niter) {
                int k1 = (i + 1) << 5;
                tdm_load_2d_bf16(ldsA0 + nxt * FFN_TILE_BYTES,
                                 (unsigned long long)(size_t)(A + (size_t)bm * K + k1),
                                 (unsigned)K, (unsigned)M, 32u, 128u, (unsigned)K);
                tdm_load_2d_bf16(ldsB0 + nxt * FFN_TILE_BYTES,
                                 (unsigned long long)(size_t)(Bt + (size_t)bn * K + k1),
                                 (unsigned)K, (unsigned)N, 32u, 128u, (unsigned)K);
                __builtin_amdgcn_s_wait_tensorcnt(2);
            } else {
                __builtin_amdgcn_s_wait_tensorcnt(0);
            }
        }
        __syncthreads();
        const __bf16* asp = &As[cur][0];
        const __bf16* bsp = &Bs[cur][0];
        FFN_ACCUM_LOOP(asp, bsp)
        __syncthreads();
    }
#else
    for (int k0 = 0; k0 < K; k0 += 32) {
#pragma unroll
        for (int i = 0; i < 2; i++) {
            int c = tid + 256 * i;
            int r = c >> 2, cc = (c & 3) * 8;
            *(v8bf*)&As[0][r * 32 + cc] = *(const v8bf*)&A[(size_t)(bm + r) * K + k0 + cc];
            *(v8bf*)&Bs[0][r * 32 + cc] = *(const v8bf*)&Bt[(size_t)(bn + r) * K + k0 + cc];
        }
        __syncthreads();
        const __bf16* asp = &As[0][0];
        const __bf16* bsp = &Bs[0][0];
        FFN_ACCUM_LOOP(asp, bsp)
        __syncthreads();
    }
#endif
    float bcol[4];
#pragma unroll
    for (int ni = 0; ni < 4; ni++) bcol[ni] = bias[bn + wn * 64 + ni * 16 + l15];
#pragma unroll
    for (int mi = 0; mi < 2; mi++)
#pragma unroll
        for (int r = 0; r < 8; r++) {
            int row = bm + wm * 32 + mi * 16 + r + hi * 8;
            float w = wgt[row];
#pragma unroll
            for (int ni = 0; ni < 4; ni++) {
                int col = bn + wn * 64 + ni * 16 + l15;
                size_t o = (size_t)row * N + col;
                Out[o] += (acc[mi][ni][r] + bcol[ni]) * w;
            }
        }
}

// ------------------------------------------------------------- small kernels
__global__ void k_router_heads(const float* __restrict__ rv,
                               const float* __restrict__ gate_w, const float* __restrict__ gate_b,
                               const float* __restrict__ tgt_w,  const float* __restrict__ tgt_b,
                               const float* __restrict__ bias,
                               float* __restrict__ logits, float* __restrict__ gs,
                               float* __restrict__ plr_out) {
    int idx = blockIdx.x * blockDim.x + threadIdx.x;
    int t = idx / 24, c = idx % 24;
    if (t >= NTOK) return;
    const float* r = rv + (size_t)t * DDIM;
    float s = 0.f;
    if (c < EEXP) {
        for (int d = 0; d < DDIM; d++) s += r[d] * gate_w[(size_t)d * EEXP + c];
        s += gate_b[c];
        logits[t * EEXP + c] = s;
        gs[t * EEXP + c] = 1.0f / (1.0f + expf(-s)) + bias[c];
    } else {
        int p = c - EEXP;
        for (int d = 0; d < DDIM; d++) s += r[d] * tgt_w[(size_t)d * 16 + p];
        plr_out[t * 16 + p] = s + tgt_b[p];
    }
}

__global__ void k_cap_head(const float* __restrict__ sact, const float* __restrict__ cap_w2,
                           const float* __restrict__ cap_b2, float* __restrict__ cap_out) {
    int idx = blockIdx.x * blockDim.x + threadIdx.x;
    if (idx >= NTOK * EEXP) return;
    int t = idx / EEXP, e = idx % EEXP;
    const float* r = sact + (size_t)t * DDIM;
    float s = 0.f;
    for (int d = 0; d < DDIM; d++) s += r[d] * cap_w2[(size_t)d * EEXP + e];
    cap_out[idx] = s + cap_b2[e];
}

// exact k-th largest of n fp32 values via 4-pass MSB radix select (1 block)
__global__ void k_topk_thr(const float* __restrict__ gs, int n, int k,
                           float* __restrict__ thr) {
    __shared__ unsigned cnt[256];
    __shared__ unsigned sh_prefix, sh_k;
    if (threadIdx.x == 0) { sh_prefix = 0u; sh_k = (unsigned)k; }
    __syncthreads();
    for (int shift = 24; shift >= 0; shift -= 8) {
        for (int i = threadIdx.x; i < 256; i += blockDim.x) cnt[i] = 0u;
        __syncthreads();
        unsigned prefix = sh_prefix;
        unsigned hmask = (shift == 24) ? 0u : (0xFFFFFFFFu << (shift + 8));
        for (int i = threadIdx.x; i < n; i += blockDim.x) {
            unsigned u = __float_as_uint(gs[i]);
            u = (u & 0x80000000u) ? ~u : (u | 0x80000000u);   // order-preserving key
            if ((u & hmask) == prefix) atomicAdd(&cnt[(u >> shift) & 0xFFu], 1u);
        }
        __syncthreads();
        if (threadIdx.x == 0) {
            unsigned kk = sh_k, cum = 0u;
            int b = 255;
            for (; b > 0; b--) {
                if (cum + cnt[b] >= kk) break;
                cum += cnt[b];
            }
            sh_k = kk - cum;
            sh_prefix = prefix | ((unsigned)b << shift);
        }
        __syncthreads();
    }
    if (threadIdx.x == 0) {
        unsigned key = sh_prefix;
        unsigned u = (key & 0x80000000u) ? (key & 0x7FFFFFFFu) : ~key;
        *thr = __uint_as_float(u);
    }
}

__global__ void k_mask_wgt(const float* __restrict__ gs, const float* __restrict__ logits,
                           const float* __restrict__ thr, float* __restrict__ mask_out,
                           float* __restrict__ wgt) {
    int idx = blockIdx.x * blockDim.x + threadIdx.x;
    if (idx >= NTOK * EEXP) return;
    int t = idx / EEXP, e = idx % EEXP;
    bool m = gs[idx] >= *thr;
    mask_out[(size_t)e * NTOK + t] = m ? 1.0f : 0.0f;
    wgt[(size_t)e * NTOK + t] = m ? logits[idx] : 0.0f;
}

// ------------------------------------------------------------------- launch
extern "C" void kernel_launch(void* const* d_in, const int* in_sizes, int n_in,
                              void* d_out, int out_size, void* d_ws, size_t ws_size,
                              hipStream_t stream) {
    const float* x        = (const float*)d_in[0];
    const float* router_w = (const float*)d_in[1];
    const float* router_b = (const float*)d_in[2];
    const float* gate_w   = (const float*)d_in[3];
    const float* gate_b   = (const float*)d_in[4];
    const float* cap_w1   = (const float*)d_in[5];
    const float* cap_b1   = (const float*)d_in[6];
    const float* cap_w2   = (const float*)d_in[7];
    const float* cap_b2   = (const float*)d_in[8];
    const float* tgt_w    = (const float*)d_in[9];
    const float* tgt_b    = (const float*)d_in[10];
    const float* bias     = (const float*)d_in[11];
    const float* ew1      = (const float*)d_in[12];
    const float* eb1      = (const float*)d_in[13];
    const float* ew2      = (const float*)d_in[14];
    const float* eb2      = (const float*)d_in[15];
    (void)in_sizes; (void)n_in; (void)out_size; (void)ws_size;

    // output layout (flat concat, reference return order)
    float* out_y    = (float*)d_out;                        // [N, D]
    float* out_mask = out_y + (size_t)NTOK * DDIM;          // [E, N]
    float* out_cap  = out_mask + (size_t)EEXP * NTOK;       // [N, E] flat
    float* out_plr  = out_cap + (size_t)NTOK * EEXP;        // [N, 16]

    // workspace layout
    char* ws = (char*)d_ws;
    size_t off = 0;
    __bf16* xbf = (__bf16*)(ws + off); off += (size_t)NTOK * DDIM * 2;   // 8 MB
    __bf16* w1t = (__bf16*)(ws + off); off += (size_t)HDIM * DDIM * 2;   // 8 MB
    __bf16* w2t = (__bf16*)(ws + off); off += (size_t)HDIM * DDIM * 2;   // 8 MB
    __bf16* hbf = (__bf16*)(ws + off); off += (size_t)NTOK * HDIM * 2;   // 32 MB
    float*  rv  = (float*)(ws + off);  off += (size_t)NTOK * DDIM * 4;   // 16 MB (reused as silu act)
    float*  logits = (float*)(ws + off); off += (size_t)NTOK * EEXP * 4;
    float*  gsbuf  = (float*)(ws + off); off += (size_t)NTOK * EEXP * 4;
    float*  wgt    = (float*)(ws + off); off += (size_t)EEXP * NTOK * 4;
    float*  thr    = (float*)(ws + off); off += 256;

    // 1) x -> bf16
    k_cvt_bf16<<<(NTOK * DDIM) / 256, 256, 0, stream>>>(x, xbf, NTOK * DDIM);

    // 2) rv = gelu(x @ router_w + b)   [exact fp32 WMMA]
    k_gemm_f32_act<0><<<dim3(DDIM / 64, NTOK / 64), 256, 0, stream>>>(
        x, router_w, router_b, rv, NTOK, DDIM, DDIM);

    // 3) logits / gates_score / plr
    k_router_heads<<<(NTOK * 24 + 255) / 256, 256, 0, stream>>>(
        rv, gate_w, gate_b, tgt_w, tgt_b, bias, logits, gsbuf, out_plr);

    // 4) s = silu(x @ cap_w1 + b)  (reuses rv buffer), then cap_pred
    k_gemm_f32_act<1><<<dim3(DDIM / 64, NTOK / 64), 256, 0, stream>>>(
        x, cap_w1, cap_b1, rv, NTOK, DDIM, DDIM);
    k_cap_head<<<(NTOK * EEXP + 255) / 256, 256, 0, stream>>>(rv, cap_w2, cap_b2, out_cap);

    // 5) global top-k threshold + mask/combine weights
    k_topk_thr<<<1, 1024, 0, stream>>>(gsbuf, NTOK * EEXP, KSEL, thr);
    k_mask_wgt<<<(NTOK * EEXP + 255) / 256, 256, 0, stream>>>(gsbuf, logits, thr, out_mask, wgt);

    // 6) zero output accumulator
    k_zero<<<(NTOK * DDIM) / 256, 256, 0, stream>>>(out_y, NTOK * DDIM);

    // 7) experts: bf16 WMMA FFN1 (TDM double-buffered, gelu-tanh) then FFN2 (+= gated)
    for (int e = 0; e < EEXP; e++) {
        k_cvtT_bf16<<<dim3(HDIM / 32, DDIM / 32), dim3(32, 8), 0, stream>>>(
            ew1 + (size_t)e * DDIM * HDIM, w1t, DDIM, HDIM);
        k_gemm_bf16_ffn1<<<dim3(HDIM / 128, NTOK / 128), 256, 0, stream>>>(
            xbf, w1t, eb1 + (size_t)e * HDIM, hbf, NTOK, HDIM, DDIM);
        k_cvtT_bf16<<<dim3(DDIM / 32, HDIM / 32), dim3(32, 8), 0, stream>>>(
            ew2 + (size_t)e * HDIM * DDIM, w2t, HDIM, DDIM);
        k_gemm_bf16_ffn2<<<dim3(DDIM / 128, NTOK / 128), 256, 0, stream>>>(
            hbf, w2t, eb2 + (size_t)e * DDIM, wgt + (size_t)e * NTOK, out_y,
            NTOK, DDIM, HDIM);
    }
}
